// HyperGATStack_12266426597833
// MI455X (gfx1250) — matrix operand order repaired and checked by reading, not compiled, not run on hardware
//
#include <hip/hip_runtime.h>
#include <hip/hip_bf16.h>

#define N_NODES 50000
#define IN_DIM  256
#define HEADS   4
#define HDIM    64
#define LAYERS  2
#define E_EDGES 20000
#define M_PAIRS 320000
#define TQ_DIM  64
#define O_TOTAL 640   // 256 Q | 256 V | 128 gate-hidden

typedef __attribute__((ext_vector_type(16))) _Float16 v16h;
typedef __attribute__((ext_vector_type(8)))  _Float16 v8h;
typedef __attribute__((ext_vector_type(8)))  float    v8f;

__device__ __forceinline__ unsigned enc_ord(float f) {
    int i = __float_as_int(f);
    return (i >= 0) ? ((unsigned)i | 0x80000000u) : (unsigned)(~i);
}
__device__ __forceinline__ float dec_ord(unsigned u) {
    int i = (u & 0x80000000u) ? (int)(u & 0x7FFFFFFFu) : ~(int)u;
    return __int_as_float(i);
}

// ---- Pack [wq | wv | w1(:, :256)] into f16, output-major [L][640][256] ----
__global__ void pack_weights(const float* __restrict__ wq, const float* __restrict__ wv,
                             const float* __restrict__ w1, _Float16* __restrict__ Wh) {
    int idx = blockIdx.x * blockDim.x + threadIdx.x;
    if (idx >= LAYERS * O_TOTAL * IN_DIM) return;
    int k = idx & 255;
    int o = (idx >> 8) % O_TOTAL;
    int l = idx / (O_TOTAL * IN_DIM);
    float v;
    if (o < 256) {
        v = wq[((l * HEADS + (o >> 6)) * HDIM + (o & 63)) * IN_DIM + k];
    } else if (o < 512) {
        int oo = o - 256;
        v = wv[((l * HEADS + (oo >> 6)) * HDIM + (oo & 63)) * IN_DIM + k];
    } else {
        int oo = o - 512;  // h = oo>>5, kk = oo&31
        v = w1[((l * HEADS + (oo >> 5)) * 32 + (oo & 31)) * (IN_DIM + TQ_DIM) + k];
    }
    Wh[idx] = (_Float16)v;
}

// ---- Gate bias per (layer,type,head,k): tq . w1[:,256:] + b1 ----
__global__ void make_tbias(const float* __restrict__ tq, const float* __restrict__ w1,
                           const float* __restrict__ b1, float* __restrict__ tb) {
    int idx = blockIdx.x * blockDim.x + threadIdx.x;
    if (idx >= LAYERS * 3 * HEADS * 32) return;
    int k = idx & 31;
    int h = (idx >> 5) & 3;
    int t = (idx >> 7) % 3;
    int l = idx / (3 * HEADS * 32);
    const float* tqp = tq + ((l * HEADS + h) * 3 + t) * TQ_DIM;
    const float* w1p = w1 + ((l * HEADS + h) * 32 + k) * (IN_DIM + TQ_DIM) + IN_DIM;
    float s = b1[(l * HEADS + h) * 32 + k];
    #pragma unroll 8
    for (int d = 0; d < TQ_DIM; ++d) s += tqp[d] * w1p[d];
    tb[idx] = s;
}

// ---- Fused WMMA GEMM: [Q|V|gate-hid] = h[N,256] x W^T[640,256] ----
// A-tile (16x256 f32) is staged into LDS as f16 ONCE per block; 8 waves then
// build their WMMA A-fragments from LDS (ds_load_b128) instead of each wave
// re-reading and re-converting the same global data (8x traffic/VALU cut).
__global__ void __launch_bounds__(256) gemm_qvh(const float* __restrict__ hin,
                                                const _Float16* __restrict__ Wl,
                                                float* __restrict__ Qb, float* __restrict__ Vb,
                                                float* __restrict__ Hb) {
    __shared__ _Float16 Atile[16 * 256];   // 8 KB of the WGP's 320 KB LDS

    const int lane   = threadIdx.x & 31;
    const int wave   = threadIdx.x >> 5;     // 8 waves, 5 o-tiles each -> O=640
    const int m_base = blockIdx.x * 16;      // 3125 blocks * 16 = 50000 exactly
    const int col    = lane & 15;
    const int kb     = (lane >> 4) * 8;      // unused; A and B both index K with kb16

    // ---- cooperative stage: each thread converts 16 contiguous floats ----
    {
        const int r = threadIdx.x >> 4;          // row 0..15
        const int c = (threadIdx.x & 15) * 16;   // col start 0..240
        const float* src = hin + (size_t)(m_base + r) * IN_DIM + c;
        const float4 f0 = *reinterpret_cast<const float4*>(src + 0);
        const float4 f1 = *reinterpret_cast<const float4*>(src + 4);
        const float4 f2 = *reinterpret_cast<const float4*>(src + 8);
        const float4 f3 = *reinterpret_cast<const float4*>(src + 12);
        v8h lo, hi;
        lo[0] = (_Float16)f0.x; lo[1] = (_Float16)f0.y; lo[2] = (_Float16)f0.z; lo[3] = (_Float16)f0.w;
        lo[4] = (_Float16)f1.x; lo[5] = (_Float16)f1.y; lo[6] = (_Float16)f1.z; lo[7] = (_Float16)f1.w;
        hi[0] = (_Float16)f2.x; hi[1] = (_Float16)f2.y; hi[2] = (_Float16)f2.z; hi[3] = (_Float16)f2.w;
        hi[4] = (_Float16)f3.x; hi[5] = (_Float16)f3.y; hi[6] = (_Float16)f3.z; hi[7] = (_Float16)f3.w;
        *reinterpret_cast<v8h*>(&Atile[r * 256 + c])     = lo;
        *reinterpret_cast<v8h*>(&Atile[r * 256 + c + 8]) = hi;
    }
    __syncthreads();

    const _Float16* arow = Atile + col * 256;
    const int kb16 = (lane >> 4) * 16;       // B: 16 consecutive K per half-wave
    v8f acc[5] = {};

    for (int ks = 0; ks < 8; ++ks) {
        const int k0 = ks * 32;
        // A fragment from LDS (contiguous K order, same as B: lane holds M=col,
        // K = kb16..kb16+15)
        const v8h a0 = *reinterpret_cast<const v8h*>(arow + k0 + kb16);
        const v8h a1 = *reinterpret_cast<const v8h*>(arow + k0 + kb16 + 8);
        v16h a;
        #pragma unroll
        for (int jj = 0; jj < 8; ++jj) { a[jj] = a0[jj]; a[8 + jj] = a1[jj]; }
        #pragma unroll
        for (int j = 0; j < 5; ++j) {
            const int o_base = (wave * 5 + j) * 16;
            // B fragment (contiguous K order, same as A: lane holds N=col, K = kb16..kb16+15)
            const _Float16* brow = Wl + (size_t)(o_base + col) * IN_DIM + k0 + kb16;
            v16h b = *reinterpret_cast<const v16h*>(brow);
            acc[j] = __builtin_amdgcn_wmma_f32_16x16x32_f16(
                false, a, false, b, (short)0, acc[j], false, false);
        }
    }

    // D layout: lane col = N, VGPR i -> M = i + 8*(lane>=16)
    const int rbase = (lane >> 4) * 8;
    #pragma unroll
    for (int j = 0; j < 5; ++j) {
        const int o_base = (wave * 5 + j) * 16;
        float* dst; int o = o_base + col; int ld;
        if (o_base < 256)      { dst = Qb; ld = 256; }
        else if (o_base < 512) { dst = Vb; o -= 256; ld = 256; }
        else                   { dst = Hb; o -= 512; ld = 128; }
        #pragma unroll
        for (int i = 0; i < 8; ++i)
            dst[(size_t)(m_base + rbase + i) * ld + o] = acc[j][i];
    }
}

// ---- Type gate: alpha_type[n,h] = sigmoid( sum_k tanh(hid + tbias) * w2 + b2 ) ----
__global__ void gate_kernel(const float* __restrict__ Hb, const int* __restrict__ node_types,
                            const float* __restrict__ tb, const float* __restrict__ w2,
                            const float* __restrict__ b2, float* __restrict__ alphaT, int l) {
    int t = blockIdx.x * blockDim.x + threadIdx.x;
    if (t >= N_NODES * HEADS) return;
    int h = t & 3, n = t >> 2;
    int ty = node_types[n];
    const float* hp  = Hb + (size_t)n * 128 + h * 32;
    const float* tbp = tb + ((l * 3 + ty) * HEADS + h) * 32;
    const float* w2p = w2 + (l * HEADS + h) * 32;
    float s = 0.f;
    #pragma unroll 8
    for (int k = 0; k < 32; ++k) s += tanhf(hp[k] + tbp[k]) * w2p[k];
    s += b2[l * HEADS + h];
    alphaT[t] = 1.f / (1.f + expf(-s));
}

__global__ void init_edge(unsigned* __restrict__ emax, float* __restrict__ eden,
                          float* __restrict__ efeat) {
    int idx = blockIdx.x * blockDim.x + threadIdx.x;
    if (idx < E_EDGES * 256) efeat[idx] = 0.f;
    if (idx < E_EDGES * HEADS) { emax[idx] = 0x007FFFFFu; /* enc(-inf) */ eden[idx] = 0.f; }
}

__global__ void zero_buf(float* __restrict__ p, int n) {
    int i = blockIdx.x * blockDim.x + threadIdx.x;
    if (i < n) p[i] = 0.f;
}

// ---- Logits + segment max ----
__global__ void alpha_max_kernel(const float* __restrict__ Qb, const float* __restrict__ ec,
                                 const float* __restrict__ alphaT, const int* __restrict__ node_idx,
                                 const int* __restrict__ edge_idx, const int* __restrict__ edge_type,
                                 float* __restrict__ attn, unsigned* __restrict__ emax, int l) {
    int t = blockIdx.x * blockDim.x + threadIdx.x;
    if (t >= M_PAIRS * HEADS) return;
    int h = t & 3, m = t >> 2;
    int n = node_idx[m], e = edge_idx[m];
    int et = edge_type[e];
    const float* q = Qb + (size_t)n * 256 + h * 64;
    const float* c = ec + ((l * HEADS + h) * 3 + et) * 64;
    float d = 0.f;
    #pragma unroll 8
    for (int i = 0; i < 64; ++i) d += q[i] * c[i];
    d = (d > 0.f) ? d : 0.2f * d;                 // leaky_relu 0.2
    d *= alphaT[(size_t)n * 4 + h];
    attn[t] = d;
    atomicMax(emax + (size_t)e * 4 + h, enc_ord(d));
}

// ---- exp(a - max) + segment sum ----
__global__ void exp_den_kernel(const int* __restrict__ edge_idx, const unsigned* __restrict__ emax,
                               float* __restrict__ attn, float* __restrict__ eden) {
    int t = blockIdx.x * blockDim.x + threadIdx.x;
    if (t >= M_PAIRS * HEADS) return;
    int h = t & 3, m = t >> 2;
    int e = edge_idx[m];
    float mx = dec_ord(emax[(size_t)e * 4 + h]);
    float ex = expf(attn[t] - mx);
    attn[t] = ex;
    atomicAdd(eden + (size_t)e * 4 + h, ex);
}

// ---- normalize + scatter attn*V into edge_feat ----
__global__ void scatter_edge_kernel(const int* __restrict__ node_idx, const int* __restrict__ edge_idx,
                                    const float* __restrict__ Vb, const float* __restrict__ eden,
                                    float* __restrict__ attn, float* __restrict__ efeat) {
    int t = blockIdx.x * blockDim.x + threadIdx.x;
    if (t >= M_PAIRS * HEADS) return;
    int h = t & 3, m = t >> 2;
    int n = node_idx[m], e = edge_idx[m];
    float a = attn[t] / eden[(size_t)e * 4 + h];
    attn[t] = a;
    const float* v = Vb + (size_t)n * 256 + h * 64;
    float* ef = efeat + (size_t)e * 256 + h * 64;
    #pragma unroll 8
    for (int i = 0; i < 64; ++i) atomicAdd(ef + i, a * v[i]);
}

// ---- gather attn*edge_feat into node accumulator ----
__global__ void gather_node_kernel(const int* __restrict__ node_idx, const int* __restrict__ edge_idx,
                                   const float* __restrict__ efeat, const float* __restrict__ attn,
                                   float* __restrict__ nacc) {
    int t = blockIdx.x * blockDim.x + threadIdx.x;
    if (t >= M_PAIRS * HEADS) return;
    int h = t & 3, m = t >> 2;
    int n = node_idx[m], e = edge_idx[m];
    float a = attn[t];
    const float* ef = efeat + (size_t)e * 256 + h * 64;
    float* na = nacc + (size_t)n * 256 + h * 64;
    #pragma unroll 8
    for (int i = 0; i < 64; ++i) atomicAdd(na + i, a * ef[i]);
}

// ---- residual + LayerNorm, one block per node ----
__global__ void __launch_bounds__(256) ln_kernel(const float* __restrict__ hin,
                                                 const float* __restrict__ nacc,
                                                 const float* __restrict__ ln_g,
                                                 const float* __restrict__ ln_b,
                                                 float* __restrict__ hout, int l) {
    __shared__ float red[256];
    int n = blockIdx.x, d = threadIdx.x;
    float v = hin[(size_t)n * 256 + d] + nacc[(size_t)n * 256 + d];
    red[d] = v; __syncthreads();
    for (int s = 128; s > 0; s >>= 1) { if (d < s) red[d] += red[d + s]; __syncthreads(); }
    float mu = red[0] * (1.f / 256.f);
    __syncthreads();
    float dv = v - mu;
    red[d] = dv * dv; __syncthreads();
    for (int s = 128; s > 0; s >>= 1) { if (d < s) red[d] += red[d + s]; __syncthreads(); }
    float var = red[0] * (1.f / 256.f);
    hout[(size_t)n * 256 + d] = dv * rsqrtf(var + 1e-5f) * ln_g[l * 256 + d] + ln_b[l * 256 + d];
}

extern "C" void kernel_launch(void* const* d_in, const int* in_sizes, int n_in,
                              void* d_out, int out_size, void* d_ws, size_t ws_size,
                              hipStream_t stream) {
    const float* x          = (const float*)d_in[0];
    const int*   node_types = (const int*)d_in[1];
    const int*   edge_type  = (const int*)d_in[2];
    const int*   node_idx   = (const int*)d_in[3];
    const int*   edge_idx   = (const int*)d_in[4];
    const float* tq         = (const float*)d_in[5];
    const float* w1         = (const float*)d_in[6];
    const float* b1         = (const float*)d_in[7];
    const float* w2         = (const float*)d_in[8];
    const float* b2         = (const float*)d_in[9];
    const float* wq         = (const float*)d_in[10];
    const float* wv         = (const float*)d_in[11];
    const float* ec         = (const float*)d_in[12];
    const float* ln_g       = (const float*)d_in[13];
    const float* ln_b       = (const float*)d_in[14];
    float* out = (float*)d_out;

    char* ws = (char*)d_ws;
    size_t off = 0;
    auto alloc = [&](size_t bytes) -> void* {
        void* p = ws + off;
        off += (bytes + 255) & ~(size_t)255;
        return p;
    };
    float*    hbuf   = (float*)alloc(sizeof(float) * (size_t)N_NODES * 256);
    float*    Qb     = (float*)alloc(sizeof(float) * (size_t)N_NODES * 256); // reused as node accumulator
    float*    Vb     = (float*)alloc(sizeof(float) * (size_t)N_NODES * 256);
    float*    Hb     = (float*)alloc(sizeof(float) * (size_t)N_NODES * 128);
    float*    alphaT = (float*)alloc(sizeof(float) * (size_t)N_NODES * HEADS);
    float*    attn   = (float*)alloc(sizeof(float) * (size_t)M_PAIRS * HEADS);
    unsigned* emax   = (unsigned*)alloc(sizeof(unsigned) * (size_t)E_EDGES * HEADS);
    float*    eden   = (float*)alloc(sizeof(float) * (size_t)E_EDGES * HEADS);
    float*    efeat  = (float*)alloc(sizeof(float) * (size_t)E_EDGES * 256);
    _Float16* Wh     = (_Float16*)alloc(sizeof(_Float16) * (size_t)LAYERS * O_TOTAL * IN_DIM);
    float*    tb     = (float*)alloc(sizeof(float) * (size_t)LAYERS * 3 * HEADS * 32);

    const int B = 256;
    pack_weights<<<(LAYERS * O_TOTAL * IN_DIM + B - 1) / B, B, 0, stream>>>(wq, wv, w1, Wh);
    make_tbias<<<(LAYERS * 3 * HEADS * 32 + B - 1) / B, B, 0, stream>>>(tq, w1, b1, tb);

    for (int l = 0; l < LAYERS; ++l) {
        const float* hin = (l == 0) ? x : hbuf;
        float* hout      = (l == LAYERS - 1) ? out : hbuf;

        gemm_qvh<<<N_NODES / 16, B, 0, stream>>>(hin, Wh + (size_t)l * O_TOTAL * IN_DIM, Qb, Vb, Hb);
        gate_kernel<<<(N_NODES * HEADS + B - 1) / B, B, 0, stream>>>(Hb, node_types, tb, w2, b2, alphaT, l);
        init_edge<<<(E_EDGES * 256 + B - 1) / B, B, 0, stream>>>(emax, eden, efeat);
        alpha_max_kernel<<<(M_PAIRS * HEADS + B - 1) / B, B, 0, stream>>>(
            Qb, ec, alphaT, node_idx, edge_idx, edge_type, attn, emax, l);
        // Q is dead now -> reuse Qb as the node accumulator
        zero_buf<<<(N_NODES * 256 + B - 1) / B, B, 0, stream>>>(Qb, N_NODES * 256);
        exp_den_kernel<<<(M_PAIRS * HEADS + B - 1) / B, B, 0, stream>>>(edge_idx, emax, attn, eden);
        scatter_edge_kernel<<<(M_PAIRS * HEADS + B - 1) / B, B, 0, stream>>>(
            node_idx, edge_idx, Vb, eden, attn, efeat);
        gather_node_kernel<<<(M_PAIRS * HEADS + B - 1) / B, B, 0, stream>>>(
            node_idx, edge_idx, efeat, attn, Qb);
        ln_kernel<<<N_NODES, B, 0, stream>>>(hin, Qb, ln_g, ln_b, hout, l);
    }
}
